// MultipathAttention_37374805410615
// MI455X (gfx1250) — matrix operand order repaired and checked by reading, not compiled, not run on hardware
//
#include <hip/hip_runtime.h>

// ---------- types ----------
typedef __attribute__((ext_vector_type(16))) __bf16         v16bf;
typedef __attribute__((ext_vector_type(8)))  float          v8f;
typedef __attribute__((ext_vector_type(16))) unsigned short v16u;
typedef __attribute__((ext_vector_type(8)))  unsigned short v8u;
typedef __attribute__((ext_vector_type(4)))  float          f4;
typedef __attribute__((ext_vector_type(4)))  int            i4;

// ---------- async-to-LDS availability (gfx1250 path) ----------
#if defined(__has_builtin)
#  if __has_builtin(__builtin_amdgcn_global_load_async_to_lds_b128)
#    define ASYNC_LDS 1
#  endif
#  if __has_builtin(__builtin_amdgcn_s_wait_asynccnt)
#    define HAVE_WAIT_ASYNC 1
#  endif
#endif
#ifndef ASYNC_LDS
#  define ASYNC_LDS 0
#endif
#ifndef HAVE_WAIT_ASYNC
#  define HAVE_WAIT_ASYNC 0
#endif

#if ASYNC_LDS
typedef __attribute__((address_space(1))) i4 gas_i4;   // int4 in global AS
typedef __attribute__((address_space(3))) i4 las_i4;   // int4 in LDS AS (32-bit ptr)

// 16B global -> LDS async copy.  AS1 ptr = 64-bit flat addr; AS3 ptr = low 32
// bits of the flat LDS-aperture address (ISA 10.2).
__device__ __forceinline__ void async_cp16(const void* g, void* l) {
  __builtin_amdgcn_global_load_async_to_lds_b128(
      (gas_i4*)(unsigned long long)g,
      (las_i4*)(unsigned)(unsigned long long)l, 0, 0);
}
#endif

__device__ __forceinline__ void wait_async0() {
#if HAVE_WAIT_ASYNC
  __builtin_amdgcn_s_wait_asynccnt(0);
#elif ASYNC_LDS
  asm volatile("s_wait_asynccnt 0x0" ::: "memory");
#endif
}

// ---------- problem constants ----------
constexpr int Lq   = 64;
constexpr int Agrp = 64;
constexpr int Sm   = 20;
constexpr int Dm   = 512;
constexpr int Hh   = 8;
constexpr int Bb   = Agrp * Sm;   // 1280
constexpr int NTOK = Lq * Bb;     // 81920

__device__ __forceinline__ unsigned short f2bf(float f) {
  unsigned u = __builtin_bit_cast(unsigned, f);
  u += 0x7FFFu + ((u >> 16) & 1u);          // round-to-nearest-even
  return (unsigned short)(u >> 16);
}

// ---------- weight transpose+convert: Wt[i*512+o] = bf16(W[o*512+i]) ----------
__global__ __launch_bounds__(256) void wconv_k(const float* __restrict__ w0, const float* __restrict__ w1,
                                               const float* __restrict__ w2, const float* __restrict__ w3,
                                               const float* __restrict__ w4, const float* __restrict__ w5,
                                               unsigned short* __restrict__ wt) {
  int idx = blockIdx.x * 256 + threadIdx.x;      // 6 * 512 * 512 total
  int wi = idx >> 18;
  int r  = idx & 262143;
  int o  = r >> 9;
  int i  = r & 511;
  const float* w = (wi == 0) ? w0 : (wi == 1) ? w1 : (wi == 2) ? w2
                 : (wi == 3) ? w3 : (wi == 4) ? w4 : w5;
  wt[(size_t)wi * 262144 + (size_t)i * 512 + o] = f2bf(w[(size_t)o * 512 + i]);
}

// ---------- generic GEMM: Y = (X[NTOK,512] @ W^T + b) * scale ----------
// LDS double-buffered W tiles (32 x 128 bf16 per k-step) shared by all 4 waves.
// MODE 0: Y -> qkv layout [bh][l][hd]  (bf16)
// MODE 1: Y -> transposed  [bh][hd][l] (bf16, for K)
// MODE 2: Y -> plain [tok][512]        (bf16)
template <int MODE, bool XF32>
__global__ __launch_bounds__(128) void gemm_k(const void* __restrict__ Xv,
                                              const unsigned short* __restrict__ Wt,
                                              const float* __restrict__ bias,
                                              unsigned short* __restrict__ Y,
                                              float scale) {
  constexpr int TW     = 128;       // tile width (u16 cols)
  constexpr int STRIDE = TW + 8;    // padded row stride -> conflict-free ds_load_b128
  constexpr int CPR    = TW / 8;    // 16B chunks per tile row
  constexpr int CPT    = TW / 32;   // chunks per thread (128 thr, 32 rows)
  __shared__ __align__(32) unsigned short Wl[2][32 * STRIDE];

  const int tid  = threadIdx.x;
  const int w    = tid >> 5;
  const int lane = tid & 31;
  const int lo   = lane & 15;
  const int hi   = lane >> 4;
  const int mrow = blockIdx.x * 64 + w * 16 + lo;   // A-frag row (token)
  const int n0   = blockIdx.y * TW;
  const int q0   = tid * CPT;

  v8u hold[CPT];                                    // fallback bounce registers
  auto issue_tile = [&](int k0, int bufidx) {
#pragma unroll
    for (int j = 0; j < CPT; ++j) {
      const int q   = q0 + j;
      const int row = q / CPR;
      const int c16 = (q % CPR) * 8;
      const unsigned short* g = Wt + (size_t)(k0 + row) * Dm + n0 + c16;
#if ASYNC_LDS
      async_cp16(g, &Wl[bufidx][row * STRIDE + c16]);
#else
      hold[j] = *(const v8u*)g;
#endif
    }
  };
  auto commit_tile = [&](int bufidx) {
#if !ASYNC_LDS
#pragma unroll
    for (int j = 0; j < CPT; ++j) {
      const int q   = q0 + j;
      const int row = q / CPR;
      const int c16 = (q % CPR) * 8;
      *(v8u*)&Wl[bufidx][row * STRIDE + c16] = hold[j];
    }
#else
    (void)bufidx;
#endif
  };

  // A raw double-buffer (converted to bf16 fragments at use time)
  f4  arf[4];
  v8u arh[2];
  auto load_a_raw = [&](int ks) {
    const int c0 = ks * 32 + hi * 8;
    if constexpr (XF32) {
      const float* xr = (const float*)Xv + (size_t)mrow * Dm;
      arf[0] = *(const f4*)(xr + c0);
      arf[1] = *(const f4*)(xr + c0 + 4);
      arf[2] = *(const f4*)(xr + c0 + 16);
      arf[3] = *(const f4*)(xr + c0 + 20);
    } else {
      const unsigned short* xr = (const unsigned short*)Xv + (size_t)mrow * Dm;
      arh[0] = *(const v8u*)(xr + c0);
      arh[1] = *(const v8u*)(xr + c0 + 16);
    }
  };
  auto conv_a = [&]() -> v16bf {
    v16u au;
    if constexpr (XF32) {
#pragma unroll
      for (int e = 0; e < 4; ++e) {
        au[e]      = f2bf(arf[0][e]);
        au[4 + e]  = f2bf(arf[1][e]);
        au[8 + e]  = f2bf(arf[2][e]);
        au[12 + e] = f2bf(arf[3][e]);
      }
    } else {
#pragma unroll
      for (int e = 0; e < 8; ++e) { au[e] = arh[0][e]; au[8 + e] = arh[1][e]; }
    }
    return __builtin_bit_cast(v16bf, au);
  };

  v8f acc[8];
#pragma unroll
  for (int t = 0; t < 8; ++t)
#pragma unroll
    for (int e = 0; e < 8; ++e) acc[t][e] = 0.f;

  // prologue: tile 0 + A raw 0
  issue_tile(0, 0);
  load_a_raw(0);
  commit_tile(0);
  wait_async0();
  __syncthreads();

  for (int ks = 0; ks < 16; ++ks) {
    const int cur = ks & 1, nxt = cur ^ 1;
    v16bf afrag = conv_a();
    if (ks < 15) {
      issue_tile((ks + 1) * 32, nxt);   // overlaps with WMMAs below
      load_a_raw(ks + 1);
    }
    const unsigned short* cb = &Wl[cur][lo];   // B-frag: lane%16 = N col, elements = K rows
#pragma unroll
    for (int nt = 0; nt < 8; ++nt) {
      v16u bu;
#pragma unroll
      for (int e = 0; e < 8; ++e) { bu[e] = cb[(hi * 8 + e) * STRIDE + nt * 16]; bu[8 + e] = cb[(16 + hi * 8 + e) * STRIDE + nt * 16]; }
      acc[nt] = __builtin_amdgcn_wmma_f32_16x16x32_bf16(
          false, afrag, false, __builtin_bit_cast(v16bf, bu), (short)0, acc[nt], false, false);
    }
    if (ks < 15) {
      commit_tile(nxt);
      wait_async0();
    }
    __syncthreads();
  }

  const int rbase = blockIdx.x * 64 + w * 16 + hi * 8;
#pragma unroll
  for (int nt = 0; nt < 8; ++nt) {
    const int col = n0 + nt * 16 + lo;
    const float bv = bias[col];
#pragma unroll
    for (int r = 0; r < 8; ++r) {
      const int tok = rbase + r;
      const float v = (acc[nt][r] + bv) * scale;
      if constexpr (MODE == 2) {
        Y[(size_t)tok * Dm + col] = f2bf(v);
      } else {
        const int l = tok / Bb, b = tok % Bb;
        const int h = col >> 6, hd = col & 63;
        const size_t bh = (size_t)b * Hh + h;
        if constexpr (MODE == 0)
          Y[(bh * 64 + l) * 64 + hd] = f2bf(v);
        else
          Y[(bh * 64 + hd) * 64 + l] = f2bf(v);
      }
    }
  }
}

// ---------- fused GLU output: out = tanh(X@Wti^T+bti) * sigmoid(X@Wtg^T+btg) ----------
__global__ __launch_bounds__(128) void gemm_glu_k(const unsigned short* __restrict__ X,
                                                  const unsigned short* __restrict__ Wti,
                                                  const unsigned short* __restrict__ Wtg,
                                                  const float* __restrict__ bti,
                                                  const float* __restrict__ btg,
                                                  float* __restrict__ out) {
  constexpr int TW     = 64;
  constexpr int STRIDE = TW + 8;    // 72 u16
  constexpr int CPR    = TW / 8;    // 8 chunks per row
  constexpr int CPT    = 4;         // 2 matrices * 256 chunks / 128 threads
  __shared__ __align__(32) unsigned short Wl[2][2][32 * STRIDE];

  const int tid  = threadIdx.x;
  const int w    = tid >> 5;
  const int lane = tid & 31;
  const int lo   = lane & 15;
  const int hi   = lane >> 4;
  const int mrow = blockIdx.x * 64 + w * 16 + lo;
  const int n0   = blockIdx.y * TW;
  const int q0   = tid * CPT;

  v8u hold[CPT];
  auto issue_tile = [&](int k0, int bufidx) {
#pragma unroll
    for (int j = 0; j < CPT; ++j) {
      const int q   = q0 + j;
      const int mat = q >> 8;           // 0 = Wti, 1 = Wtg
      const int rem = q & 255;
      const int row = rem / CPR;
      const int c16 = (rem % CPR) * 8;
      const unsigned short* g = (mat ? Wtg : Wti) + (size_t)(k0 + row) * Dm + n0 + c16;
#if ASYNC_LDS
      async_cp16(g, &Wl[bufidx][mat][row * STRIDE + c16]);
#else
      hold[j] = *(const v8u*)g;
#endif
    }
  };
  auto commit_tile = [&](int bufidx) {
#if !ASYNC_LDS
#pragma unroll
    for (int j = 0; j < CPT; ++j) {
      const int q   = q0 + j;
      const int mat = q >> 8;
      const int rem = q & 255;
      const int row = rem / CPR;
      const int c16 = (rem % CPR) * 8;
      *(v8u*)&Wl[bufidx][mat][row * STRIDE + c16] = hold[j];
    }
#else
    (void)bufidx;
#endif
  };

  v8u arh[2];
  auto load_a_raw = [&](int ks) {
    const int c0 = ks * 32 + hi * 8;
    const unsigned short* xr = X + (size_t)mrow * Dm;
    arh[0] = *(const v8u*)(xr + c0);
    arh[1] = *(const v8u*)(xr + c0 + 16);
  };

  v8f ai[4], ag[4];
#pragma unroll
  for (int t = 0; t < 4; ++t)
#pragma unroll
    for (int e = 0; e < 8; ++e) { ai[t][e] = 0.f; ag[t][e] = 0.f; }

  issue_tile(0, 0);
  load_a_raw(0);
  commit_tile(0);
  wait_async0();
  __syncthreads();

  for (int ks = 0; ks < 16; ++ks) {
    const int cur = ks & 1, nxt = cur ^ 1;
    v16u au;
#pragma unroll
    for (int e = 0; e < 8; ++e) { au[e] = arh[0][e]; au[8 + e] = arh[1][e]; }
    v16bf afrag = __builtin_bit_cast(v16bf, au);
    if (ks < 15) {
      issue_tile((ks + 1) * 32, nxt);
      load_a_raw(ks + 1);
    }
    const unsigned short* ci = &Wl[cur][0][lo];
    const unsigned short* cg = &Wl[cur][1][lo];
#pragma unroll
    for (int nt = 0; nt < 4; ++nt) {
      v16u bi;
#pragma unroll
      for (int e = 0; e < 8; ++e) { bi[e] = ci[(hi * 8 + e) * STRIDE + nt * 16]; bi[8 + e] = ci[(16 + hi * 8 + e) * STRIDE + nt * 16]; }
      ai[nt] = __builtin_amdgcn_wmma_f32_16x16x32_bf16(
          false, afrag, false, __builtin_bit_cast(v16bf, bi), (short)0, ai[nt], false, false);
      v16u bg;
#pragma unroll
      for (int e = 0; e < 8; ++e) { bg[e] = cg[(hi * 8 + e) * STRIDE + nt * 16]; bg[8 + e] = cg[(16 + hi * 8 + e) * STRIDE + nt * 16]; }
      ag[nt] = __builtin_amdgcn_wmma_f32_16x16x32_bf16(
          false, afrag, false, __builtin_bit_cast(v16bf, bg), (short)0, ag[nt], false, false);
    }
    if (ks < 15) {
      commit_tile(nxt);
      wait_async0();
    }
    __syncthreads();
  }

  const int rbase = blockIdx.x * 64 + w * 16 + hi * 8;
#pragma unroll
  for (int nt = 0; nt < 4; ++nt) {
    const int col = n0 + nt * 16 + lo;
    const float bv_i = bti[col];
    const float bv_g = btg[col];
#pragma unroll
    for (int r = 0; r < 8; ++r) {
      const float yi = tanhf(ai[nt][r] + bv_i);
      const float yg = 1.f / (1.f + __expf(-(ag[nt][r] + bv_g)));
      out[(size_t)(rbase + r) * Dm + col] = yi * yg;
    }
  }
}

// ---------- per-(b,h) attention: S = Q K^T (+mask), softmax, O = P V ----------
__global__ __launch_bounds__(128) void attn_k(const unsigned short* __restrict__ Q,
                                              const unsigned short* __restrict__ KT,
                                              const unsigned short* __restrict__ V,
                                              const float* __restrict__ amask,
                                              unsigned short* __restrict__ O) {
  __shared__ __align__(32) unsigned short P[4 * 16 * 64];   // per-wave P tiles (bf16)

  const int bh   = blockIdx.x;
  const int b    = bh >> 3;
  const int h    = bh & 7;
  const int w    = threadIdx.x >> 5;
  const int lane = threadIdx.x & 31;
  const int lo   = lane & 15;
  const int hi   = lane >> 4;
  const int m0   = w * 16;                 // query-row tile for this wave

  const unsigned short* Qb = Q  + (size_t)bh * 4096;
  const unsigned short* Kb = KT + (size_t)bh * 4096;
  const unsigned short* Vb = V  + (size_t)bh * 4096;

  // --- Q A-fragments (two 32-wide K steps) ---
  v16bf qa[2];
#pragma unroll
  for (int ks = 0; ks < 2; ++ks) {
    const int c0 = ks * 32 + hi * 8;
    const unsigned short* qr = Qb + (size_t)(m0 + lo) * 64;
    v8u a0 = *(const v8u*)(qr + c0);
    v8u a1 = *(const v8u*)(qr + c0 + 16);
    v16u au;
#pragma unroll
    for (int e = 0; e < 8; ++e) { au[e] = a0[e]; au[8 + e] = a1[e]; }
    qa[ks] = __builtin_bit_cast(v16bf, au);
  }

  // --- scores: 4 n-tiles x 2 k-steps of WMMA ---
  v8f sc[4];
#pragma unroll
  for (int t = 0; t < 4; ++t)
#pragma unroll
    for (int e = 0; e < 8; ++e) sc[t][e] = 0.f;
#pragma unroll
  for (int nt = 0; nt < 4; ++nt)
#pragma unroll
    for (int ks = 0; ks < 2; ++ks) {
      const int krow = ks * 32 + hi * 8;          // B-frag: lane%16 = key col, elements = hd rows of K^T
      v16u bu;
#pragma unroll
      for (int e = 0; e < 8; ++e) { bu[e] = Kb[(size_t)(krow + e) * 64 + nt * 16 + lo]; bu[8 + e] = Kb[(size_t)(krow + 16 + e) * 64 + nt * 16 + lo]; }
      sc[nt] = __builtin_amdgcn_wmma_f32_16x16x32_bf16(
          false, qa[ks], false, __builtin_bit_cast(v16bf, bu), (short)0, sc[nt], false, false);
    }

  // --- causal mask + attn_mask + row softmax (rows live across 16 lanes) ---
  const int ibase = m0 + hi * 8;
#pragma unroll
  for (int r = 0; r < 8; ++r) {
    const int i = ibase + r;
#pragma unroll
    for (int nt = 0; nt < 4; ++nt) {
      const int j = nt * 16 + lo;
      float v = sc[nt][r] + amask[i * 64 + j];
      if (j > i) v = -1e30f;                      // seq_mask causal
      sc[nt][r] = v;
    }
    float mx = fmaxf(fmaxf(sc[0][r], sc[1][r]), fmaxf(sc[2][r], sc[3][r]));
#pragma unroll
    for (int msk = 1; msk <= 8; msk <<= 1) mx = fmaxf(mx, __shfl_xor(mx, msk, 32));
    float s = 0.f;
#pragma unroll
    for (int nt = 0; nt < 4; ++nt) {
      const float e = __expf(sc[nt][r] - mx);
      sc[nt][r] = e;
      s += e;
    }
#pragma unroll
    for (int msk = 1; msk <= 8; msk <<= 1) s += __shfl_xor(s, msk, 32);
    const float inv = 1.f / s;
#pragma unroll
    for (int nt = 0; nt < 4; ++nt) sc[nt][r] *= inv;
  }

  // --- P -> LDS (row-major bf16), reload as A-fragments ---
#pragma unroll
  for (int nt = 0; nt < 4; ++nt)
#pragma unroll
    for (int r = 0; r < 8; ++r)
      P[w * 1024 + (hi * 8 + r) * 64 + nt * 16 + lo] = f2bf(sc[nt][r]);
  __syncthreads();

  v16bf pa[2];
#pragma unroll
  for (int ks = 0; ks < 2; ++ks) {
    const int c0 = ks * 32 + hi * 8;
    const unsigned short* pr = &P[w * 1024 + lo * 64];
    v8u a0 = *(const v8u*)(pr + c0);
    v8u a1 = *(const v8u*)(pr + c0 + 16);
    v16u au;
#pragma unroll
    for (int e = 0; e < 8; ++e) { au[e] = a0[e]; au[8 + e] = a1[e]; }
    pa[ks] = __builtin_bit_cast(v16bf, au);
  }

  // --- O = P @ V ---
  v8f oa[4];
#pragma unroll
  for (int t = 0; t < 4; ++t)
#pragma unroll
    for (int e = 0; e < 8; ++e) oa[t][e] = 0.f;
#pragma unroll
  for (int nt = 0; nt < 4; ++nt)
#pragma unroll
    for (int ks = 0; ks < 2; ++ks) {
      const int krow = ks * 32 + hi * 8;          // B-frag: lane%16 = hd col, elements = key positions
      v16u bu;
#pragma unroll
      for (int e = 0; e < 8; ++e) { bu[e] = Vb[(size_t)(krow + e) * 64 + nt * 16 + lo]; bu[8 + e] = Vb[(size_t)(krow + 16 + e) * 64 + nt * 16 + lo]; }
      oa[nt] = __builtin_amdgcn_wmma_f32_16x16x32_bf16(
          false, pa[ks], false, __builtin_bit_cast(v16bf, bu), (short)0, oa[nt], false, false);
    }

  // --- write O back in [token][512] bf16 layout for Wo GEMM ---
#pragma unroll
  for (int nt = 0; nt < 4; ++nt)
#pragma unroll
    for (int r = 0; r < 8; ++r) {
      const int l  = m0 + hi * 8 + r;
      const int hd = nt * 16 + lo;
      O[((size_t)l * Bb + b) * Dm + h * 64 + hd] = f2bf(oa[nt][r]);
    }
}

// ---------- host launch ----------
extern "C" void kernel_launch(void* const* d_in, const int* in_sizes, int n_in,
                              void* d_out, int out_size, void* d_ws, size_t ws_size,
                              hipStream_t stream) {
  (void)in_sizes; (void)n_in; (void)out_size; (void)ws_size;

  const float* query = (const float*)d_in[0];
  const float* key_  = (const float*)d_in[1];
  const float* value = (const float*)d_in[2];
  const float* amask = (const float*)d_in[3];
  const float* Wq  = (const float*)d_in[4];
  const float* bq  = (const float*)d_in[5];
  const float* Wk  = (const float*)d_in[6];
  const float* bk  = (const float*)d_in[7];
  const float* Wv  = (const float*)d_in[8];
  const float* bv  = (const float*)d_in[9];
  const float* Wo  = (const float*)d_in[10];
  const float* bo  = (const float*)d_in[11];
  const float* Wti = (const float*)d_in[12];
  const float* bti = (const float*)d_in[13];
  const float* Wtg = (const float*)d_in[14];
  const float* btg = (const float*)d_in[15];

  // workspace layout (u16 elements)
  unsigned short* wt    = (unsigned short*)d_ws;          // 6 * 512*512 bf16 transposed weights
  const size_t WSTEP    = (size_t)512 * 512;              // 262144
  const size_t TOKSZ    = (size_t)NTOK * Dm;              // 41,943,040 elems (~84 MB)
  unsigned short* qws   = wt + 6 * WSTEP;                 // Q  [bh][l][hd]
  unsigned short* ktws  = qws + TOKSZ;                    // K^T[bh][hd][l]
  unsigned short* vws   = ktws + TOKSZ;                   // V  [bh][l][hd]
  unsigned short* attnw = vws + TOKSZ;                    // attn output [tok][512]
  unsigned short* tattn = qws;                            // reuse Q region for t_attn

  const dim3 blk(128);
  const dim3 gW((6 * 512 * 512) / 256);
  const dim3 gG(NTOK / 64, 4);
  const dim3 gGlu(NTOK / 64, 8);

  wconv_k<<<gW, 256, 0, stream>>>(Wq, Wk, Wv, Wo, Wti, Wtg, wt);

  gemm_k<0, true><<<gG, blk, 0, stream>>>(query, wt + 0 * WSTEP, bq, qws, 0.125f);  // 1/sqrt(64)
  gemm_k<1, true><<<gG, blk, 0, stream>>>(key_,  wt + 1 * WSTEP, bk, ktws, 1.0f);
  gemm_k<0, true><<<gG, blk, 0, stream>>>(value, wt + 2 * WSTEP, bv, vws, 1.0f);

  attn_k<<<Bb * Hh, blk, 0, stream>>>(qws, ktws, vws, amask, attnw);

  gemm_k<2, false><<<gG, blk, 0, stream>>>(attnw, wt + 3 * WSTEP, bo, tattn, 1.0f);

  gemm_glu_k<<<gGlu, blk, 0, stream>>>(tattn, wt + 4 * WSTEP, wt + 5 * WSTEP, bti, btg,
                                       (float*)d_out);
}
